// GraphAttentionLayer_48318382080392
// MI455X (gfx1250) — matrix operand order repaired and checked by reading, not compiled, not run on hardware
//
#include <hip/hip_runtime.h>

typedef __attribute__((ext_vector_type(16))) __bf16 v16bf;
typedef __attribute__((ext_vector_type(2)))  __bf16 v2bf;
typedef __attribute__((ext_vector_type(8)))  float  v8f;

constexpr int NN  = 4096;
constexpr int FIN = 128;
constexpr int HD  = 512;   // H*D
constexpr int NH  = 4;
#define L2E 1.44269504088896340736f

// raw v_exp_f32 (exp2)
__device__ __forceinline__ float hw_exp2(float x) {
  return __builtin_amdgcn_exp2f(x);
}

// native bf16 converts -> hardware v_cvt (RNE), instead of manual bit-twiddling
__device__ __forceinline__ unsigned pk2(float a, float b) {
  v2bf t;
  t[0] = (__bf16)a;
  t[1] = (__bf16)b;
  return __builtin_bit_cast(unsigned, t);
}
__device__ __forceinline__ unsigned short f2bf(float x) {
  return __builtin_bit_cast(unsigned short, (__bf16)x);
}
__device__ __forceinline__ float bf2f(unsigned short s) {
  return __builtin_bit_cast(float, ((unsigned)s) << 16);
}

union Vb { unsigned u[8]; v16bf v; };

// ---------------------------------------------------------------------------
// Kernel 1: ht = h @ W  (4096x128 * 128x512), bf16 WMMA, f32 accumulate.
// Stores htT[col][row] in bf16 (feature-major transpose) so the attention
// kernel's B-operand loads are contiguous 32B per lane.
// ---------------------------------------------------------------------------
__global__ __launch_bounds__(128) void gat_proj(const float* __restrict__ h,
                                                const float* __restrict__ W,
                                                unsigned short* __restrict__ htT) {
  const int tid = threadIdx.x, wv = tid >> 5, lane = tid & 31;
  const int hf = lane >> 4, rl = lane & 15;
  const int tile = blockIdx.x * 4 + wv;          // 0..8191
  const int m0 = (tile >> 5) * 16, n0 = (tile & 31) * 16;
  const float* hrow = h + (size_t)(m0 + rl) * FIN;
  const int c = n0 + rl;                         // global output column
  v8f acc = {};
#pragma unroll
  for (int kc = 0; kc < FIN; kc += 32) {
    // A (16x32 bf16): lane<16 holds K {0..7,16..23}, lane>=16 holds {8..15,24..31}
    const int kb = kc + 8 * hf;
    float4 f0 = *(const float4*)(hrow + kb);
    float4 f1 = *(const float4*)(hrow + kb + 4);
    float4 f2 = *(const float4*)(hrow + kb + 16);
    float4 f3 = *(const float4*)(hrow + kb + 20);
    Vb A;
    A.u[0] = pk2(f0.x, f0.y); A.u[1] = pk2(f0.z, f0.w);
    A.u[2] = pk2(f1.x, f1.y); A.u[3] = pk2(f1.z, f1.w);
    A.u[4] = pk2(f2.x, f2.y); A.u[5] = pk2(f2.z, f2.w);
    A.u[6] = pk2(f3.x, f3.y); A.u[7] = pk2(f3.z, f3.w);
    // B (32x16 bf16): lane<16 col=lane K {0..7,16..23}, lane>=16 col=lane-16 K {8..15,24..31}
    const int kbb = kc + 8 * hf;
    float b[16];
#pragma unroll
    for (int kk = 0; kk < 16; kk++) b[kk] = W[(size_t)(kbb + kk + (kk & 8)) * HD + c];
    Vb B;
#pragma unroll
    for (int i = 0; i < 8; i++) B.u[i] = pk2(b[2 * i], b[2 * i + 1]);
    acc = __builtin_amdgcn_wmma_f32_16x16x32_bf16(false, A.v, false, B.v,
                                                  (short)0, acc, false, false);
  }
  // C layout: vgpr v -> row v (lanes 0-15) / v+8 (lanes 16-31), col = lane&15
#pragma unroll
  for (int v = 0; v < 8; v++) {
    int row = m0 + v + 8 * hf;
    htT[(size_t)c * NN + row] = f2bf(acc[v]);
  }
}

// ---------------------------------------------------------------------------
// Kernel 2: src[h,i] = ht[h,i,:].a[h,:D], tgt[h,i] = ht[h,i,:].a[h,D:]
// (outputs pre-scaled by log2(e) so the attention kernel can use exp2)
// ---------------------------------------------------------------------------
__global__ __launch_bounds__(128) void gat_srctgt(const unsigned short* __restrict__ htT,
                                                  const float* __restrict__ a,
                                                  float* __restrict__ src,
                                                  float* __restrict__ tgt) {
  const int head = blockIdx.x >> 5;                       // uniform per block
  const int row = (blockIdx.x * 128 + threadIdx.x) & (NN - 1);
  const float* ah = a + head * 256;
  float s = 0.f, t = 0.f;
  for (int d = 0; d < 128; d++) {
    float x = bf2f(htT[(size_t)(head * 128 + d) * NN + row]);
    s = fmaf(x, ah[d], s);
    t = fmaf(x, ah[128 + d], t);
  }
  src[head * NN + row] = s * L2E;
  tgt[head * NN + row] = t * L2E;
}

// ---------------------------------------------------------------------------
// Kernel 3: fused flash attention + head-mean + LayerNorm + residual + pos.
// 1 block = 16 query rows; 4 waves = 4 heads; j-tiles of 32 keys.
// All logits are in log2(e) units -> softmax via v_exp_f32 (exp2) directly.
// ---------------------------------------------------------------------------
__global__ __launch_bounds__(128) void gat_attn(
    const float* __restrict__ hin, const int* __restrict__ adj,
    const float* __restrict__ pos, const unsigned short* __restrict__ htT,
    const float* __restrict__ srcg, const float* __restrict__ tgtg,
    const float* __restrict__ gamma, const float* __restrict__ beta,
    float* __restrict__ hout, float* __restrict__ posout) {
  __shared__ float s_posi[48];
  __shared__ float s_src[NH][16];
  __shared__ float s_posj[96];
  __shared__ float s_tgt[NH][32];
  __shared__ float s_dist[16][32];
  __shared__ float s_adj[16][32];
  __shared__ alignas(16) float s_scale[NH][16];
  __shared__ alignas(16) unsigned s_pst[NH][320];   // 16 rows x 20-dword stride
  __shared__ float s_hacc[NH][16][128];
  __shared__ float s_pacc[NH][16][3];
  __shared__ float s_part[16][8][2];

  const int tid = threadIdx.x, wv = tid >> 5, lane = tid & 31;
  const int hf = lane >> 4, rl = lane & 15, cb = hf * 16;
  const int i0 = blockIdx.x * 16;

  if (tid < 48) s_posi[tid] = pos[i0 * 3 + tid];
  if (tid < 64) s_src[tid >> 4][tid & 15] = srcg[(tid >> 4) * NN + i0 + (tid & 15)];
  __syncthreads();

  v8f acc[8];
#pragma unroll
  for (int t = 0; t < 8; t++) acc[t] = {};
  float m_run = -3e38f, l_run = 0.f, dax = 0.f, day = 0.f, daz = 0.f;

  for (int j0 = 0; j0 < NN; j0 += 32) {
    // ---- stage 1: cooperative loads -------------------------------------
    if (tid < 96) s_posj[tid] = pos[j0 * 3 + tid];
    s_tgt[wv][lane] = tgtg[wv * NN + j0 + lane];
    {
      int rr = tid >> 3, c0 = (tid & 7) * 4;
      int4 av = *(const int4*)(adj + (size_t)(i0 + rr) * NN + j0 + c0);
      s_adj[rr][c0 + 0] = av.x > 0 ? 1.f : 0.f;
      s_adj[rr][c0 + 1] = av.y > 0 ? 1.f : 0.f;
      s_adj[rr][c0 + 2] = av.z > 0 ? 1.f : 0.f;
      s_adj[rr][c0 + 3] = av.w > 0 ? 1.f : 0.f;
    }
    __syncthreads();
    // ---- stage 2: pairwise distances (shared by all 4 heads) ------------
    // pre-scaled by log2(e): LeakyReLU/mask commute with positive scaling
#pragma unroll
    for (int q = 0; q < 4; q++) {
      int idx = tid * 4 + q, rr = idx >> 5, cc = idx & 31;
      float dx = s_posi[rr * 3 + 0] - s_posj[cc * 3 + 0];
      float dy = s_posi[rr * 3 + 1] - s_posj[cc * 3 + 1];
      float dz = s_posi[rr * 3 + 2] - s_posj[cc * 3 + 2];
      s_dist[rr][cc] = sqrtf(dx * dx + dy * dy + dz * dz) * L2E;
    }
    __syncthreads();
    // ---- stage 3: per-head logits + online softmax ----------------------
    // stats layout: lane owns row rl, columns cb..cb+15
    float sr = s_src[wv][rl];
    float e[16];
#pragma unroll
    for (int c2 = 0; c2 < 16; c2++) {
      int col = cb + c2;
      float x = sr + s_tgt[wv][col] - s_dist[rl][col];
      x = x > 0.f ? x : 0.2f * x;                        // LeakyReLU(0.2)
      float mk = s_adj[rl][col];
      e[c2] = fmaf(x, mk, fmaf(1.3e16f, mk, -1.3e16f));  // mask -> -9e15*log2e
    }
    float ml = e[0];
#pragma unroll
    for (int c2 = 1; c2 < 16; c2++) ml = fmaxf(ml, e[c2]);
    float mo = __shfl_xor(ml, 16, 32);
    float m_new = fmaxf(fmaxf(ml, mo), m_run);
    float sc = hw_exp2(m_run - m_new);
    m_run = m_new;
    float p[16], ps = 0.f;
#pragma unroll
    for (int c2 = 0; c2 < 16; c2++) { p[c2] = hw_exp2(e[c2] - m_new); ps += p[c2]; }
    float po = __shfl_xor(ps, 16, 32);
    l_run = fmaf(l_run, sc, ps + po);
    // P @ pos_j for delta (f32 VALU, same rescaling)
    float ax = 0.f, ay = 0.f, az = 0.f;
#pragma unroll
    for (int c2 = 0; c2 < 16; c2++) {
      int col = cb + c2;
      ax = fmaf(p[c2], s_posj[col * 3 + 0], ax);
      ay = fmaf(p[c2], s_posj[col * 3 + 1], ay);
      az = fmaf(p[c2], s_posj[col * 3 + 2], az);
    }
    dax = fmaf(dax, sc, ax); day = fmaf(day, sc, ay); daz = fmaf(daz, sc, az);
    // rescale O accumulators: broadcast per-row scale into C layout
    if (hf == 0) s_scale[wv][rl] = sc;      // same-wave DS ops are in-order
    float4 s0 = *(const float4*)&s_scale[wv][hf * 8];
    float4 s1 = *(const float4*)&s_scale[wv][hf * 8 + 4];
    v8f sv = {s0.x, s0.y, s0.z, s0.w, s1.x, s1.y, s1.z, s1.w};
#pragma unroll
    for (int t = 0; t < 8; t++) acc[t] *= sv;
    // stage bf16 P through LDS into the WMMA 16-bit A-matrix layout
    unsigned* prow = &s_pst[wv][rl * 20];
    *(uint4*)(prow + hf * 8) =
        make_uint4(pk2(p[0], p[1]), pk2(p[2], p[3]), pk2(p[4], p[5]), pk2(p[6], p[7]));
    *(uint4*)(prow + hf * 8 + 4) =
        make_uint4(pk2(p[8], p[9]), pk2(p[10], p[11]), pk2(p[12], p[13]), pk2(p[14], p[15]));
    Vb A;
    *(uint4*)&A.u[0] = *(const uint4*)(prow + hf * 8);       // K 0..7 / 16..23
    *(uint4*)&A.u[4] = *(const uint4*)(prow + hf * 8 + 4);   // K 8..15 / 24..31
    // B tiles from bf16 htT (contiguous 32B per lane) + WMMA accumulate
    const unsigned short* hb = htT + (size_t)(wv * 128) * NN + j0 + hf * 16;
#pragma unroll
    for (int t = 0; t < 8; t++) {
      v16bf B = *(const v16bf*)(hb + (size_t)(t * 16 + rl) * NN);
      acc[t] = __builtin_amdgcn_wmma_f32_16x16x32_bf16(false, A.v, false, B,
                                                       (short)0, acc[t], false, false);
    }
    __syncthreads();
  }

  // ---- finalize: 1/l scaling, per-head outputs to LDS ----------------------
  float inv = 1.0f / l_run;
  float fdx = dax + __shfl_xor(dax, 16, 32);
  float fdy = day + __shfl_xor(day, 16, 32);
  float fdz = daz + __shfl_xor(daz, 16, 32);
  if (hf == 0) {
    s_pacc[wv][rl][0] = s_posi[rl * 3 + 0] - fdx * inv;   // delta_head (rowsum=1)
    s_pacc[wv][rl][1] = s_posi[rl * 3 + 1] - fdy * inv;
    s_pacc[wv][rl][2] = s_posi[rl * 3 + 2] - fdz * inv;
    s_scale[wv][rl] = inv;
  }
  float4 s0 = *(const float4*)&s_scale[wv][hf * 8];
  float4 s1 = *(const float4*)&s_scale[wv][hf * 8 + 4];
  float iv[8] = {s0.x, s0.y, s0.z, s0.w, s1.x, s1.y, s1.z, s1.w};
#pragma unroll
  for (int t = 0; t < 8; t++)
#pragma unroll
    for (int v = 0; v < 8; v++)
      s_hacc[wv][v + 8 * hf][t * 16 + rl] = acc[t][v] * iv[v];
  __syncthreads();

  // ---- head mean + LayerNorm + residual (deterministic reductions) --------
  {
    int r = tid >> 3, d0 = (tid & 7) * 16;
    float sum = 0.f, sq = 0.f;
#pragma unroll
    for (int i = 0; i < 16; i++) {
      int d = d0 + i;
      float x = 0.25f * (s_hacc[0][r][d] + s_hacc[1][r][d] + s_hacc[2][r][d] + s_hacc[3][r][d]);
      sum += x; sq += x * x;
    }
    s_part[r][tid & 7][0] = sum;
    s_part[r][tid & 7][1] = sq;
    __syncthreads();
    float ts = 0.f, tq = 0.f;
#pragma unroll
    for (int i = 0; i < 8; i++) { ts += s_part[r][i][0]; tq += s_part[r][i][1]; }
    float mu = ts * (1.f / 128.f);
    float var = tq * (1.f / 128.f) - mu * mu;
    float rstd = rsqrtf(var + 1e-5f);
#pragma unroll
    for (int i = 0; i < 16; i++) {
      int d = d0 + i;
      float x = 0.25f * (s_hacc[0][r][d] + s_hacc[1][r][d] + s_hacc[2][r][d] + s_hacc[3][r][d]);
      float hn = (x - mu) * rstd * gamma[d] + beta[d];
      hout[(size_t)(i0 + r) * 128 + d] = hin[(size_t)(i0 + r) * 128 + d] + hn;
    }
  }
  if (tid < 48) {
    int r = tid / 3, k = tid % 3;
    float dsum = 0.25f * (s_pacc[0][r][k] + s_pacc[1][r][k] + s_pacc[2][r][k] + s_pacc[3][r][k]);
    posout[(i0 + r) * 3 + k] = pos[(i0 + r) * 3 + k] + dsum;
  }
}

// ---------------------------------------------------------------------------
extern "C" void kernel_launch(void* const* d_in, const int* in_sizes, int n_in,
                              void* d_out, int out_size, void* d_ws, size_t ws_size,
                              hipStream_t stream) {
  const float* h     = (const float*)d_in[0];
  const int*   adj   = (const int*)d_in[1];
  const float* pos   = (const float*)d_in[2];
  const float* W     = (const float*)d_in[3];
  const float* a     = (const float*)d_in[4];
  const float* gamma = (const float*)d_in[5];
  const float* beta  = (const float*)d_in[6];

  unsigned short* htT = (unsigned short*)d_ws;                       // 4 MB bf16
  float* src = (float*)((char*)d_ws + (size_t)HD * NN * sizeof(unsigned short));
  float* tgt = src + (size_t)NH * NN;

  float* hout   = (float*)d_out;
  float* posout = hout + (size_t)NN * 128;

  gat_proj<<<2048, 128, 0, stream>>>(h, W, htT);
  gat_srctgt<<<128, 128, 0, stream>>>(htT, a, src, tgt);
  gat_attn<<<256, 128, 0, stream>>>(h, adj, pos, htT, src, tgt, gamma, beta,
                                    hout, posout);
}
